// GNN_1_74242804678665
// MI455X (gfx1250) — compile-verified, matrix operand order checked by reading, not run on hardware
//
#include <hip/hip_runtime.h>
#include <hip/hip_bf16.h>

typedef __bf16 v16bf  __attribute__((ext_vector_type(16)));
typedef __bf16 bf16x8 __attribute__((ext_vector_type(8)));
typedef float  v8f    __attribute__((ext_vector_type(8)));

#define DD  256
#define D2  512
#define FIN 40
#define NL  5

// ---------- small utility kernels ----------
__global__ void k_zero(float* p, int n) {
    int i = blockIdx.x * blockDim.x + threadIdx.x;
    if (i < n) p[i] = 0.0f;
}

__global__ void k_cvt_bf16(const float* __restrict__ s, __bf16* __restrict__ d, int n) {
    int i = blockIdx.x * blockDim.x + threadIdx.x;
    if (i < n) d[i] = (__bf16)s[i];
}

// W (K x Nc, row-major fp32) -> Wt (Nc x K, row-major bf16)
__global__ void k_transpose_cvt(const float* __restrict__ W, __bf16* __restrict__ Wt,
                                int K, int Nc) {
    int i = blockIdx.x * blockDim.x + threadIdx.x;
    if (i >= K * Nc) return;
    int k = i / Nc, n = i % Nc;
    Wt[n * K + k] = (__bf16)W[i];
}

// ---------- embedding: h = relu(x @ embW + embB) ----------
__global__ void k_embed(const float* __restrict__ x, const float* __restrict__ embW,
                        const float* __restrict__ embB, float* __restrict__ h, int N) {
    __shared__ float xs[FIN];
    int n = blockIdx.x;
    int t = threadIdx.x;
    if (t < FIN) xs[t] = x[n * FIN + t];
    __syncthreads();
    float acc = embB[t];
    #pragma unroll 8
    for (int k = 0; k < FIN; ++k) acc = fmaf(xs[k], embW[k * DD + t], acc);
    h[n * DD + t] = acc > 0.0f ? acc : 0.0f;
}

// ---------- agg = h + ee1[l][4] + ee2[l][0] ----------
__global__ void k_init_agg(const float* __restrict__ h, const float* __restrict__ e1,
                           const float* __restrict__ e2, float* __restrict__ agg, int total) {
    int i = blockIdx.x * blockDim.x + threadIdx.x;
    if (i >= total) return;
    int d = i & (DD - 1);
    agg[i] = h[i] + e1[d] + e2[d];
}

// ---------- message scatter-add: agg[dst] += h[src] + ee1[ea0] + ee2[ea1] ----------
__global__ void k_scatter(const float* __restrict__ h, const int* __restrict__ src,
                          const int* __restrict__ dst, const int* __restrict__ eattr,
                          const float* __restrict__ ee1l, const float* __restrict__ ee2l,
                          float* __restrict__ agg) {
    int e = blockIdx.x;           // one edge per block (uniform scalar loads)
    int d = threadIdx.x;          // 256 threads = D features
    int s  = src[e];
    int t  = dst[e];
    int a0 = eattr[e * 2 + 0];
    int a1 = eattr[e * 2 + 1];
    float v = h[s * DD + d] + ee1l[a0 * DD + d] + ee2l[a1 * DD + d];
    unsafeAtomicAdd(&agg[t * DD + d], v);
}

// ---------- WMMA fragment load (A row-major, Bt pre-transposed => same layout) ----------
// 16-bit A/B 16x32 layout: lanes 0-15 hold K=[0..7]+[16..23]; lanes 16-31 K=[8..15]+[24..31].
__device__ __forceinline__ v16bf load_frag(const __bf16* __restrict__ base, int stride,
                                           int row0, int k0, int lane) {
    int r    = row0 + (lane & 15);
    int koff = k0 + ((lane >> 4) << 3);
    union { v16bf v; bf16x8 h[2]; } u;
    u.h[0] = *(const bf16x8*)(base + (size_t)r * stride + koff);
    u.h[1] = *(const bf16x8*)(base + (size_t)r * stride + koff + 16);
    return u.v;
}

#define WMMA_BF16(a, b, c) \
    __builtin_amdgcn_wmma_f32_16x16x32_bf16(false, (a), false, (b), (short)0, (c), false, false)

// ---------- GEMM1: D(bf16) = relu(A(bf16) @ B + bias) ; 32x32 tile per wave ----------
__global__ void k_gemm_bias_relu_bf16(const __bf16* __restrict__ A, const __bf16* __restrict__ Bt,
                                      const float* __restrict__ bias, __bf16* __restrict__ D,
                                      int M, int K, int Nc) {
    int lane = threadIdx.x & 31;
    int wave = threadIdx.x >> 5;
    int colTiles = Nc >> 5;                                  // 32-wide col tiles
    int tile = blockIdx.x * (blockDim.x >> 5) + wave;
    if (tile >= (M >> 5) * colTiles) return;
    int row0 = (tile / colTiles) << 5;
    int col0 = (tile % colTiles) << 5;
    v8f c00 = {0.f,0.f,0.f,0.f,0.f,0.f,0.f,0.f};
    v8f c01 = c00, c10 = c00, c11 = c00;
    for (int k = 0; k < K; k += 32) {
        v16bf a0 = load_frag(A,  K, row0,      k, lane);
        v16bf a1 = load_frag(A,  K, row0 + 16, k, lane);
        v16bf b0 = load_frag(Bt, K, col0,      k, lane);
        v16bf b1 = load_frag(Bt, K, col0 + 16, k, lane);
        c00 = WMMA_BF16(a0, b0, c00);
        c01 = WMMA_BF16(a0, b1, c01);
        c10 = WMMA_BF16(a1, b0, c10);
        c11 = WMMA_BF16(a1, b1, c11);
    }
    // C/D layout: VGPR i -> M = i (+8 for lanes 16-31), N = lane & 15
    int n0 = col0 + (lane & 15), n1 = n0 + 16;
    float bv0 = bias[n0], bv1 = bias[n1];
    int m0 = row0 + ((lane >> 4) << 3), m1 = m0 + 16;
    #pragma unroll
    for (int i = 0; i < 8; ++i) {
        float v;
        v = c00[i] + bv0; v = v > 0.f ? v : 0.f; D[(size_t)(m0 + i) * Nc + n0] = (__bf16)v;
        v = c01[i] + bv1; v = v > 0.f ? v : 0.f; D[(size_t)(m0 + i) * Nc + n1] = (__bf16)v;
        v = c10[i] + bv0; v = v > 0.f ? v : 0.f; D[(size_t)(m1 + i) * Nc + n0] = (__bf16)v;
        v = c11[i] + bv1; v = v > 0.f ? v : 0.f; D[(size_t)(m1 + i) * Nc + n1] = (__bf16)v;
    }
}

// ---------- GEMM2: D(f32) = A(bf16) @ B + bias ; 32x32 tile/wave ; fused BN partial sums ----------
__global__ void k_gemm_bias_f32_bnstats(const __bf16* __restrict__ A, const __bf16* __restrict__ Bt,
                                        const float* __restrict__ bias, float* __restrict__ D,
                                        float* __restrict__ sums, int M, int K, int Nc) {
    int lane = threadIdx.x & 31;
    int wave = threadIdx.x >> 5;
    int colTiles = Nc >> 5;
    int tile = blockIdx.x * (blockDim.x >> 5) + wave;
    if (tile >= (M >> 5) * colTiles) return;
    int row0 = (tile / colTiles) << 5;
    int col0 = (tile % colTiles) << 5;
    v8f c00 = {0.f,0.f,0.f,0.f,0.f,0.f,0.f,0.f};
    v8f c01 = c00, c10 = c00, c11 = c00;
    for (int k = 0; k < K; k += 32) {
        v16bf a0 = load_frag(A,  K, row0,      k, lane);
        v16bf a1 = load_frag(A,  K, row0 + 16, k, lane);
        v16bf b0 = load_frag(Bt, K, col0,      k, lane);
        v16bf b1 = load_frag(Bt, K, col0 + 16, k, lane);
        c00 = WMMA_BF16(a0, b0, c00);
        c01 = WMMA_BF16(a0, b1, c01);
        c10 = WMMA_BF16(a1, b0, c10);
        c11 = WMMA_BF16(a1, b1, c11);
    }
    int n0 = col0 + (lane & 15), n1 = n0 + 16;
    float bv0 = bias[n0], bv1 = bias[n1];
    int m0 = row0 + ((lane >> 4) << 3), m1 = m0 + 16;
    float s0 = 0.f, q0 = 0.f, s1 = 0.f, q1 = 0.f;   // per-column partial sum / sumsq
    #pragma unroll
    for (int i = 0; i < 8; ++i) {
        float v;
        v = c00[i] + bv0; D[(size_t)(m0 + i) * Nc + n0] = v; s0 += v; q0 += v * v;
        v = c10[i] + bv0; D[(size_t)(m1 + i) * Nc + n0] = v; s0 += v; q0 += v * v;
        v = c01[i] + bv1; D[(size_t)(m0 + i) * Nc + n1] = v; s1 += v; q1 += v * v;
        v = c11[i] + bv1; D[(size_t)(m1 + i) * Nc + n1] = v; s1 += v; q1 += v * v;
    }
    unsafeAtomicAdd(&sums[n0], s0);
    unsafeAtomicAdd(&sums[DD + n0], q0);
    unsafeAtomicAdd(&sums[n1], s1);
    unsafeAtomicAdd(&sums[DD + n1], q1);
}

__global__ void k_bn_apply(const float* __restrict__ h2, const float* __restrict__ sums,
                           const float* __restrict__ g, const float* __restrict__ b,
                           float* __restrict__ h, int N, int doRelu) {
    int i = blockIdx.x * blockDim.x + threadIdx.x;
    if (i >= N * DD) return;
    int d = i & (DD - 1);
    float invN = 1.0f / (float)N;
    float mu  = sums[d] * invN;
    float var = sums[DD + d] * invN - mu * mu;
    float v = (h2[i] - mu) * rsqrtf(var + 1e-5f) * g[d] + b[d];
    if (doRelu) v = v > 0.0f ? v : 0.0f;
    h[i] = v;
}

// ---------- graph pooling ----------
__global__ void k_pool(const float* __restrict__ h, const int* __restrict__ batch,
                       float* __restrict__ mol, int* __restrict__ cnt) {
    int n = blockIdx.x;
    int d = threadIdx.x;
    int g = batch[n];
    unsafeAtomicAdd(&mol[(size_t)g * DD + d], h[(size_t)n * DD + d]);
    if (d == 0) atomicAdd(&cnt[g], 1);
}

// ---------- per-graph MLP head: 256->256->128->64->32->1 ----------
__global__ void k_head(const float* __restrict__ mol, const int* __restrict__ cnt,
                       const float* W0, const float* B0, const float* W1h, const float* B1,
                       const float* W2h, const float* B2, const float* W3, const float* B3,
                       const float* Wo, const float* Bo, float* __restrict__ out) {
    __shared__ float in[DD], z1[DD], z2[128], z3[64], z4[32];
    int g = blockIdx.x, t = threadIdx.x;
    int c = cnt[g]; if (c < 1) c = 1;
    in[t] = mol[(size_t)g * DD + t] / (float)c;
    __syncthreads();
    { float a = B0[t]; for (int k = 0; k < DD; ++k) a = fmaf(in[k], W0[k * DD + t], a);
      z1[t] = a > 0.f ? a : 0.f; }
    __syncthreads();
    if (t < 128) { float a = B1[t]; for (int k = 0; k < DD; ++k) a = fmaf(z1[k], W1h[k * 128 + t], a);
      z2[t] = a > 0.f ? a : 0.f; }
    __syncthreads();
    if (t < 64) { float a = B2[t]; for (int k = 0; k < 128; ++k) a = fmaf(z2[k], W2h[k * 64 + t], a);
      z3[t] = a > 0.f ? a : 0.f; }
    __syncthreads();
    if (t < 32) { float a = B3[t]; for (int k = 0; k < 64; ++k) a = fmaf(z3[k], W3[k * 32 + t], a);
      z4[t] = a > 0.f ? a : 0.f; }
    __syncthreads();
    if (t == 0) { float a = Bo[0]; for (int k = 0; k < 32; ++k) a = fmaf(z4[k], Wo[k], a);
      out[g] = a; }
}

extern "C" void kernel_launch(void* const* d_in, const int* in_sizes, int n_in,
                              void* d_out, int out_size, void* d_ws, size_t ws_size,
                              hipStream_t stream) {
    const float* x     = (const float*)d_in[0];
    const int*   eidx  = (const int*)  d_in[1];
    const int*   eattr = (const int*)  d_in[2];
    const int*   batch = (const int*)  d_in[3];
    const float* embW  = (const float*)d_in[4];
    const float* embB  = (const float*)d_in[5];
    const float* ee1   = (const float*)d_in[6];
    const float* ee2   = (const float*)d_in[7];
    const float* W1    = (const float*)d_in[8];
    const float* b1    = (const float*)d_in[9];
    const float* W2    = (const float*)d_in[10];
    const float* b2    = (const float*)d_in[11];
    const float* bng   = (const float*)d_in[12];
    const float* bnb   = (const float*)d_in[13];
    const float* hW0 = (const float*)d_in[14]; const float* hb0 = (const float*)d_in[15];
    const float* hW1 = (const float*)d_in[16]; const float* hb1 = (const float*)d_in[17];
    const float* hW2 = (const float*)d_in[18]; const float* hb2 = (const float*)d_in[19];
    const float* hW3 = (const float*)d_in[20]; const float* hb3 = (const float*)d_in[21];
    const float* hWo = (const float*)d_in[22]; const float* hbo = (const float*)d_in[23];

    const int N = in_sizes[0] / FIN;
    const int E = in_sizes[1] / 2;
    const int G = out_size;
    const int* src = eidx;
    const int* dst = eidx + E;

    // workspace carve-out (256B aligned)
    char* base = (char*)d_ws; size_t off = 0;
    auto alloc = [&](size_t bytes) -> char* {
        char* p = base + off;
        off += (bytes + 255) & ~(size_t)255;
        return p;
    };
    float*  h    = (float*) alloc((size_t)N * DD * 4);
    float*  agg  = (float*) alloc((size_t)N * DD * 4);   // reused as h2 (GEMM2 output)
    __bf16* aggb = (__bf16*)alloc((size_t)N * DD * 2);
    __bf16* midb = (__bf16*)alloc((size_t)N * D2 * 2);
    __bf16* w1t  = (__bf16*)alloc((size_t)NL * DD * D2 * 2);
    __bf16* w2t  = (__bf16*)alloc((size_t)NL * DD * D2 * 2);
    float*  sums = (float*) alloc(2 * DD * 4);
    float*  mol  = (float*) alloc((size_t)G * DD * 4);
    int*    cnt  = (int*)   alloc((size_t)G * 4);

    // pre-transpose + convert weights to bf16 (L2-resident afterwards)
    for (int l = 0; l < NL; ++l) {
        k_transpose_cvt<<<(DD * D2 + 255) / 256, 256, 0, stream>>>(
            W1 + (size_t)l * DD * D2, w1t + (size_t)l * DD * D2, DD, D2);
        k_transpose_cvt<<<(DD * D2 + 255) / 256, 256, 0, stream>>>(
            W2 + (size_t)l * D2 * DD, w2t + (size_t)l * D2 * DD, D2, DD);
    }

    k_embed<<<N, DD, 0, stream>>>(x, embW, embB, h, N);

    const int totND = N * DD;
    for (int l = 0; l < NL; ++l) {
        const float* ee1l = ee1 + (size_t)l * 6 * DD;
        const float* ee2l = ee2 + (size_t)l * 3 * DD;
        k_init_agg<<<(totND + 255) / 256, 256, 0, stream>>>(h, ee1l + 4 * DD, ee2l, agg, totND);
        k_scatter<<<E, DD, 0, stream>>>(h, src, dst, eattr, ee1l, ee2l, agg);
        k_cvt_bf16<<<(totND + 255) / 256, 256, 0, stream>>>(agg, aggb, totND);

        int tiles1 = (N / 32) * (D2 / 32);
        k_gemm_bias_relu_bf16<<<(tiles1 + 7) / 8, 256, 0, stream>>>(
            aggb, w1t + (size_t)l * DD * D2, b1 + (size_t)l * D2, midb, N, DD, D2);

        k_zero<<<2, 256, 0, stream>>>(sums, 2 * DD);
        int tiles2 = (N / 32) * (DD / 32);
        k_gemm_bias_f32_bnstats<<<(tiles2 + 7) / 8, 256, 0, stream>>>(
            midb, w2t + (size_t)l * DD * D2, b2 + (size_t)l * DD, agg, sums, N, D2, DD);

        k_bn_apply<<<(totND + 255) / 256, 256, 0, stream>>>(
            agg, sums, bng + (size_t)l * DD, bnb + (size_t)l * DD, h, N, (l < NL - 1) ? 1 : 0);
    }

    k_zero<<<(G * DD + 255) / 256, 256, 0, stream>>>(mol, G * DD);
    k_zero<<<(G + 255) / 256, 256, 0, stream>>>((float*)cnt, G);
    k_pool<<<N, DD, 0, stream>>>(h, batch, mol, cnt);
    k_head<<<G, 256, 0, stream>>>(mol, cnt, hW0, hb0, hW1, hb1, hW2, hb2, hW3, hb3,
                                  hWo, hbo, (float*)d_out);
}
